// CrossAttentionTemporal3D_76227079569561
// MI455X (gfx1250) — matrix operand order repaired and checked by reading, not compiled, not run on hardware
//
#include <hip/hip_runtime.h>
#include <cstdint>

typedef __bf16 bf16;
typedef __attribute__((ext_vector_type(16))) __bf16 v16bf;
typedef __attribute__((ext_vector_type(8)))  __bf16 v8bf;
typedef __attribute__((ext_vector_type(8)))  float  v8f;
typedef __attribute__((ext_vector_type(4)))  unsigned int u32x4;
typedef __attribute__((ext_vector_type(8)))  int i32x8;
typedef __attribute__((ext_vector_type(4)))  int i32x4;

constexpr int B_ = 2, S_ = 576, F_ = 16, D_ = 512, H_ = 8, C_ = 64;
constexpr int NTOK = B_ * S_ * F_;   // 18432 tokens, dim 512
constexpr int DF_  = 9;              // frames kept for frame-0 query attention
constexpr int NKQ  = S_ * DF_;       // 5184 keys per (b,h)

#if defined(__gfx1250__) && __has_builtin(__builtin_amdgcn_tensor_load_to_lds) && \
    __has_builtin(__builtin_amdgcn_s_wait_tensorcnt)
#define USE_TDM 1
#else
#define USE_TDM 0
#endif

__device__ __forceinline__ v8f zero8() {
    v8f z;
#pragma unroll
    for (int i = 0; i < 8; i++) z[i] = 0.0f;
    return z;
}

// A-fragment (16x32 bf16): lane L holds row M=L%16; element e maps to
// K = 2*h8 + e, h8 = (L>=16 ? 8 : 0). Two 16B runs per 32-K chunk.
__device__ __forceinline__ v16bf load_afrag(const bf16* rowbase, int koff, int h8) {
    v8bf lo = *(const v8bf*)(rowbase + koff + 2 * h8);
    v8bf hi = *(const v8bf*)(rowbase + koff + 2 * h8 + 8);
    v16bf r;
#pragma unroll
    for (int i = 0; i < 8; i++) { r[i] = lo[i]; r[8 + i] = hi[i]; }
    return r;
}

// B-fragment (32x16 bf16): lane L holds column N=L%16; element e maps to
// K = k0 + e with k0 = (L>=16 ? 16 : 0) (+ chunk offset). Needs K-contiguous rows.
__device__ __forceinline__ v16bf load_bfrag(const bf16* rowbase, int k0) {
    v8bf lo = *(const v8bf*)(rowbase + k0);
    v8bf hi = *(const v8bf*)(rowbase + k0 + 8);
    v16bf r;
#pragma unroll
    for (int i = 0; i < 8; i++) { r[i] = lo[i]; r[8 + i] = hi[i]; }
    return r;
}

__device__ __forceinline__ v8f wmma_bf16(v16bf a, v16bf b, v8f c) {
    return __builtin_amdgcn_wmma_f32_16x16x32_bf16(false, a, false, b, (short)0, c, false, false);
}

// ---------------------------------------------------------------------------
// f32 -> bf16 conversion (weights)
// ---------------------------------------------------------------------------
__global__ void cvt_f32_bf16(const float* __restrict__ s, bf16* __restrict__ d, int n) {
    int i = blockIdx.x * 256 + threadIdx.x;
    if (i < n) d[i] = (bf16)s[i];
}

// ---------------------------------------------------------------------------
// Build kept-frame table from drop_mask: fsel[b] = {0,1, 2+i for dm[b][i]==0}
// ---------------------------------------------------------------------------
__global__ void build_fsel(const int* __restrict__ dm, int* __restrict__ fsel) {
    int b = threadIdx.x;
    if (b >= B_) return;
    int idx = 0;
    fsel[b * 16 + idx++] = 0;
    fsel[b * 16 + idx++] = 1;
    for (int i = 0; i < F_ - 2; i++)
        if (dm[b * (F_ - 2) + i] == 0 && idx < 16) fsel[b * 16 + idx++] = 2 + i;
    for (; idx < 16; idx++) fsel[b * 16 + idx] = 0;
}

// ---------------------------------------------------------------------------
// Tiled GEMM: (NTOK x 512) @ (512 x 512). 128x128 block tile, K-steps of 32,
// 256 threads = 8 waves, each wave computes a 16x128 strip (8 WMMA acc tiles).
// INBF:   A is bf16 (else f32 converted on the fly)
// OUTF32: write f32 + bias (else write bf16)
// ---------------------------------------------------------------------------
template <bool INBF, bool OUTF32>
__global__ void gemm_k(const void* __restrict__ Aptr, const bf16* __restrict__ Bw,
                       const float* __restrict__ bias, void* __restrict__ Out) {
    __shared__ __attribute__((aligned(16))) bf16 ldsA[128][40]; // [m][k], pad 8
    __shared__ __attribute__((aligned(16))) bf16 ldsB[128][40]; // [n][k], pad 8

    const int mt = blockIdx.x, nt = blockIdx.y;
    const int tid = threadIdx.x;
    const int wid = tid >> 5;
    const int lane = tid & 31;
    const int ln = lane & 15;
    const bool hl = lane >= 16;
    const int h8 = hl ? 8 : 0;
    const int k0b = hl ? 16 : 0;

    v8f acc[8];
#pragma unroll
    for (int ns = 0; ns < 8; ns++) acc[ns] = zero8();

    for (int kt = 0; kt < 16; kt++) {
        // ---- prefetch next K-step tiles into cache hierarchy ----
        if (kt + 1 < 16) {
            int ktn = kt + 1;
            if constexpr (!INBF) {
                const float* A = (const float*)Aptr;
                __builtin_prefetch(A + (size_t)(mt * 128 + (tid >> 3)) * 512 + ktn * 32 + (tid & 7) * 4, 0, 0);
            } else {
                const bf16* A = (const bf16*)Aptr;
                __builtin_prefetch(A + (size_t)(mt * 128 + (tid >> 2)) * 512 + ktn * 32 + (tid & 3) * 8, 0, 0);
            }
            __builtin_prefetch(Bw + (size_t)(ktn * 32 + (tid >> 4)) * 512 + nt * 128 + (tid & 15) * 8, 0, 0);
        }
        // ---- stage A tile (128 x 32) ----
        if constexpr (!INBF) {
            const float* A = (const float*)Aptr;
#pragma unroll
            for (int i = 0; i < 4; i++) {
                int id = tid + i * 256;           // 0..1023 float4 units
                int row = id >> 3, c4 = id & 7;
                float4 f = *(const float4*)(A + (size_t)(mt * 128 + row) * 512 + kt * 32 + c4 * 4);
                bf16* d = &ldsA[row][c4 * 4];
                d[0] = (bf16)f.x; d[1] = (bf16)f.y; d[2] = (bf16)f.z; d[3] = (bf16)f.w;
            }
        } else {
            const bf16* A = (const bf16*)Aptr;
#pragma unroll
            for (int i = 0; i < 2; i++) {
                int id = tid + i * 256;           // 0..511 v8bf units
                int row = id >> 2, c8 = id & 3;
                v8bf vv = *(const v8bf*)(A + (size_t)(mt * 128 + row) * 512 + kt * 32 + c8 * 8);
                *(v8bf*)&ldsA[row][c8 * 8] = vv;
            }
        }
        // ---- stage B tile (32 x 128), transposed into [n][k] ----
#pragma unroll
        for (int i = 0; i < 2; i++) {
            int id = tid + i * 256;               // 0..511 v8bf units
            int kr = id >> 4, n8 = id & 15;
            v8bf vv = *(const v8bf*)(Bw + (size_t)(kt * 32 + kr) * 512 + nt * 128 + n8 * 8);
#pragma unroll
            for (int j = 0; j < 8; j++) ldsB[n8 * 8 + j][kr] = vv[j];
        }
        __syncthreads();

        v16bf afrag = load_afrag(&ldsA[wid * 16 + ln][0], 0, h8);
#pragma unroll
        for (int ns = 0; ns < 8; ns++) {
            v16bf bfr = load_bfrag(&ldsB[ns * 16 + ln][0], k0b);
            acc[ns] = wmma_bf16(afrag, bfr, acc[ns]);
        }
        __syncthreads();
    }

    // ---- store: C/D layout row = r + 8*(lane>=16), col = lane%16 ----
    const int ro = hl ? 8 : 0;
#pragma unroll
    for (int ns = 0; ns < 8; ns++) {
        int col = nt * 128 + ns * 16 + ln;
        if constexpr (OUTF32) {
            float bcol = bias[col];
            float* O = (float*)Out;
#pragma unroll
            for (int r = 0; r < 8; r++) {
                int row = mt * 128 + wid * 16 + ro + r;
                O[(size_t)row * 512 + col] = acc[ns][r] + bcol;
            }
        } else {
            bf16* O = (bf16*)Out;
#pragma unroll
            for (int r = 0; r < 8; r++) {
                int row = mt * 128 + wid * 16 + ro + r;
                O[(size_t)row * 512 + col] = (bf16)acc[ns][r];
            }
        }
    }
}

// ---------------------------------------------------------------------------
// Flash attention.  128 threads = 4 waves, each wave owns 16 query rows of a
// 64-query block; loop over 64-key blocks with online softmax.
// MODE 0: spatial attention per (b, f in 1..15, h), 576 keys (same frame).
//         K tile is fetched by the Tensor Data Mover (one tensor_load_to_lds
//         per 64-key block, pad-to-72-element rows), waited on TENSORcnt.
// MODE 1: frame-0 query attention per (b, h), 5184 keys = (s, fsel[j]) pairs
//         (non-affine gather -> manual staging).
// ---------------------------------------------------------------------------
template <int MODE>
__global__ void attn_k(const bf16* __restrict__ qb, const bf16* __restrict__ kb,
                       const bf16* __restrict__ vb, bf16* __restrict__ res,
                       const int* __restrict__ fsel) {
    __shared__ __attribute__((aligned(16))) bf16 ldsK[64][72];      // [key][c]
    __shared__ __attribute__((aligned(16))) bf16 ldsVt[64][72];     // [c][key]
    __shared__ __attribute__((aligned(16))) bf16 ldsP[4][16][72];   // per-wave P

    const int qblk = blockIdx.x;
    int b, f, h, NK;
    if constexpr (MODE == 0) {
        int idx = blockIdx.y;
        b = idx / ((F_ - 1) * H_);
        int rem = idx % ((F_ - 1) * H_);
        f = 1 + rem / H_;
        h = rem % H_;
        NK = S_;
    } else {
        b = blockIdx.y / H_;
        h = blockIdx.y % H_;
        f = 0;
        NK = NKQ;
    }

    const int tid = threadIdx.x;
    const int wid = tid >> 5;
    const int lane = tid & 31;
    const int ln = lane & 15;
    const bool hl = lane >= 16;
    const int h8 = hl ? 8 : 0;
    const int k0b = hl ? 16 : 0;

    // Q fragments for this wave's 16 rows, straight from global (bf16)
    const int srow = qblk * 64 + wid * 16 + ln;
    const bf16* qrow = qb + (((size_t)(b * S_ + srow) * F_ + f) * 512 + h * 64);
    v16bf qf0 = load_afrag(qrow, 0, h8);
    v16bf qf1 = load_afrag(qrow, 32, h8);

    float m[8], l[8];
    v8f acc[4];
#pragma unroll
    for (int r = 0; r < 8; r++) { m[r] = -3.0e38f; l[r] = 0.0f; }
#pragma unroll
    for (int cs = 0; cs < 4; cs++) acc[cs] = zero8();

    const int nkb = NK / 64;
    for (int kblk = 0; kblk < nkb; kblk++) {
        // ---- stage K (row-major) and V (transposed) tiles ----
        {
            int t = tid >> 1;                   // key 0..63
            int c0 = (tid & 1) * 32;            // c half
            size_t koff;
            if constexpr (MODE == 0) {
                int sk = kblk * 64 + t;
                koff = ((size_t)(b * S_ + sk) * F_ + f) * 512 + h * 64 + c0;
            } else {
                int kk = kblk * 64 + t;
                int sk = kk / DF_;
                int j = kk - sk * DF_;
                int fj = fsel[b * 16 + j];
                koff = ((size_t)(b * S_ + sk) * F_ + fj) * 512 + h * 64 + c0;
            }

            bool manual_k = true;
#if USE_TDM
            if constexpr (MODE == 0) {
                manual_k = false;
                if (wid == 0) {
                    // --- Tensor DMA: 64 rows x 128B, row stride F*D*2 bytes,
                    //     LDS rows padded to 144B (pad 4 DW per 32 DW). ---
                    uint64_t ga = (uint64_t)(uintptr_t)(kb +
                        ((size_t)((b * S_ + kblk * 64) * F_ + f) * 512 + h * 64));
                    uint32_t laddr = (uint32_t)(uintptr_t)(void*)&ldsK[0][0];
                    u32x4 g0;
                    g0[0] = 1u;                                   // count=1 valid descriptor
                    g0[1] = laddr;                                // lds_addr (bytes)
                    g0[2] = (uint32_t)ga;                         // global_addr[31:0]
                    g0[3] = (uint32_t)((ga >> 32) & 0x1FFFFFFu)   // global_addr[56:32]
                          | (2u << 30);                           // type=2 (image)
                    i32x8 g1;
                    g1[0] = (int)((1u << 16)      // data_size = 2 bytes
                                | (1u << 20)      // pad_enable
                                | (4u << 22)      // pad_interval: 32 DWORDs (128B row)
                                | (3u << 25));    // pad_amount: 4 DWORDs (8 bf16)
                    g1[1] = (int)(64u << 16);     // tensor_dim0[15:0] = 64
                    g1[2] = (int)(64u << 16);     // tensor_dim1[15:0] = 64
                    g1[3] = (int)(64u << 16);     // tile_dim0 = 64
                    g1[4] = 64;                   // tile_dim1 = 64, tile_dim2 = 0
                    g1[5] = F_ * 512;             // tensor_dim0_stride = 8192 elems
                    g1[6] = 0;
                    g1[7] = 0;
                    i32x4 gz = {0, 0, 0, 0};
#if __clang_major__ >= 23
                    i32x8 gz8 = {0, 0, 0, 0, 0, 0, 0, 0};
                    __builtin_amdgcn_tensor_load_to_lds(g0, g1, gz, gz, gz8, 0);
#else
                    __builtin_amdgcn_tensor_load_to_lds(g0, g1, gz, gz, 0);
#endif
                }
            }
#endif
            if (manual_k) {
#pragma unroll
                for (int w = 0; w < 4; w++) {
                    v8bf kv = *(const v8bf*)(kb + koff + w * 8);
                    *(v8bf*)&ldsK[t][c0 + w * 8] = kv;
                }
            }
#pragma unroll
            for (int w = 0; w < 4; w++) {
                v8bf vv = *(const v8bf*)(vb + koff + w * 8);
#pragma unroll
                for (int j = 0; j < 8; j++) ldsVt[c0 + w * 8 + j][t] = vv[j];
            }
#if USE_TDM
            if constexpr (MODE == 0) {
                if (wid == 0) __builtin_amdgcn_s_wait_tensorcnt(0);
            }
#endif
        }
        __syncthreads();

        // ---- logits S = scale * Q K^T : 4 key-subtiles of 16 ----
        v8f s4[4];
#pragma unroll
        for (int ktile = 0; ktile < 4; ktile++) {
            v8f s = zero8();
            v16bf b0 = load_bfrag(&ldsK[ktile * 16 + ln][0], k0b);
            s = wmma_bf16(qf0, b0, s);
            v16bf b1 = load_bfrag(&ldsK[ktile * 16 + ln][0], 32 + k0b);
            s = wmma_bf16(qf1, b1, s);
            s4[ktile] = s * 0.125f;             // 1/sqrt(64)
        }

        // ---- online softmax (rows live in 16-lane halves) ----
#pragma unroll
        for (int r = 0; r < 8; r++) {
            float mx = fmaxf(fmaxf(s4[0][r], s4[1][r]), fmaxf(s4[2][r], s4[3][r]));
#pragma unroll
            for (int msk = 1; msk < 16; msk <<= 1) mx = fmaxf(mx, __shfl_xor(mx, msk, 32));
            float mnew = fmaxf(m[r], mx);
            float alpha = __expf(m[r] - mnew);
            float rs = 0.0f;
#pragma unroll
            for (int ktile = 0; ktile < 4; ktile++) {
                float p = __expf(s4[ktile][r] - mnew);
                rs += p;
                ldsP[wid][r + (hl ? 8 : 0)][ktile * 16 + ln] = (bf16)p;
            }
#pragma unroll
            for (int msk = 1; msk < 16; msk <<= 1) rs += __shfl_xor(rs, msk, 32);
            l[r] = l[r] * alpha + rs;
            m[r] = mnew;
#pragma unroll
            for (int cs = 0; cs < 4; cs++) acc[cs][r] = acc[cs][r] * alpha;
        }
        __syncthreads();   // P staged; also keeps K/V tile stable until PV done

        // ---- O += P V ----
#pragma unroll
        for (int chunk = 0; chunk < 2; chunk++) {
            v16bf pf = load_afrag(&ldsP[wid][ln][0], chunk * 32, h8);
#pragma unroll
            for (int cs = 0; cs < 4; cs++) {
                v16bf vf = load_bfrag(&ldsVt[cs * 16 + ln][0], chunk * 32 + k0b);
                acc[cs] = wmma_bf16(pf, vf, acc[cs]);
            }
        }
        __syncthreads();   // before next block overwrites ldsK/ldsVt
    }

    // ---- normalize and store to res (bf16) ----
    const int ro = hl ? 8 : 0;
#pragma unroll
    for (int cs = 0; cs < 4; cs++) {
        int col = h * 64 + cs * 16 + ln;
#pragma unroll
        for (int r = 0; r < 8; r++) {
            int sq = qblk * 64 + wid * 16 + ro + r;
            res[((size_t)(b * S_ + sq) * F_ + f) * 512 + col] = (bf16)(acc[cs][r] / l[r]);
        }
    }
}

// ---------------------------------------------------------------------------
extern "C" void kernel_launch(void* const* d_in, const int* in_sizes, int n_in,
                              void* d_out, int out_size, void* d_ws, size_t ws_size,
                              hipStream_t stream) {
    (void)in_sizes; (void)n_in; (void)out_size; (void)ws_size;
    const float* x    = (const float*)d_in[0];
    const int*   dm   = (const int*)d_in[1];
    const float* Wq   = (const float*)d_in[2];
    const float* Wk   = (const float*)d_in[3];
    const float* Wv   = (const float*)d_in[4];
    const float* Wout = (const float*)d_in[5];
    const float* bout = (const float*)d_in[6];

    char* ws = (char*)d_ws;
    constexpr size_t WSZ = (size_t)512 * 512 * 2;     // bf16 weight matrix
    constexpr size_t TSZ = (size_t)NTOK * 512 * 2;    // bf16 token tensor
    bf16* wqb   = (bf16*)(ws);
    bf16* wkb   = (bf16*)(ws + WSZ);
    bf16* wvb   = (bf16*)(ws + 2 * WSZ);
    bf16* woutb = (bf16*)(ws + 3 * WSZ);
    int*  fsel  = (int*)(ws + 4 * WSZ);
    char* big   = ws + 4 * WSZ + 4096;
    bf16* qbuf  = (bf16*)(big);
    bf16* kbuf  = (bf16*)(big + TSZ);
    bf16* vbuf  = (bf16*)(big + 2 * TSZ);
    bf16* resb  = (bf16*)(big + 3 * TSZ);

    const int nW = 512 * 512;
    cvt_f32_bf16<<<(nW + 255) / 256, 256, 0, stream>>>(Wq, wqb, nW);
    cvt_f32_bf16<<<(nW + 255) / 256, 256, 0, stream>>>(Wk, wkb, nW);
    cvt_f32_bf16<<<(nW + 255) / 256, 256, 0, stream>>>(Wv, wvb, nW);
    cvt_f32_bf16<<<(nW + 255) / 256, 256, 0, stream>>>(Wout, woutb, nW);
    build_fsel<<<1, 32, 0, stream>>>(dm, fsel);

    dim3 gg(NTOK / 128, 512 / 128);
    gemm_k<false, false><<<gg, 256, 0, stream>>>(x, wqb, nullptr, qbuf);
    gemm_k<false, false><<<gg, 256, 0, stream>>>(x, wkb, nullptr, kbuf);
    gemm_k<false, false><<<gg, 256, 0, stream>>>(x, wvb, nullptr, vbuf);

    attn_k<0><<<dim3(S_ / 64, B_ * (F_ - 1) * H_), 128, 0, stream>>>(qbuf, kbuf, vbuf, resb, fsel);
    attn_k<1><<<dim3(S_ / 64, B_ * H_),            128, 0, stream>>>(qbuf, kbuf, vbuf, resb, fsel);

    gemm_k<true, true><<<gg, 256, 0, stream>>>(resb, woutb, bout, d_out);
}
